// Asc_GridCrossAttention_77721728189028
// MI455X (gfx1250) — compile-verified, matrix operand order checked by reading, not run on hardware
//
#include <hip/hip_runtime.h>

typedef float  v8f   __attribute__((ext_vector_type(8)));
typedef __bf16 v16bf __attribute__((ext_vector_type(16)));
typedef unsigned int u32x4 __attribute__((ext_vector_type(4)));
typedef int i32x8 __attribute__((ext_vector_type(8)));
typedef int i32x4 __attribute__((ext_vector_type(4)));

constexpr int C   = 128;
constexpr int H   = 4;
constexpr int HD  = 32;
constexpr int NKV = 12288;
constexpr int S   = 12288;
constexpr int NH  = 9;
constexpr int T   = 2;
constexpr int NQ  = S * 16;
constexpr int AS  = 136;   // padded LDS row stride for A tiles (bf16 elems)

// ---- CDNA5 split dependency counters ----
#define LDSF()    asm volatile("s_wait_dscnt 0x0" ::: "memory")
#define AWAIT0()  asm volatile("s_wait_asynccnt 0x0" ::: "memory")
// partial wait: allow N async ops still outstanding
#define AWAITN(n) asm volatile("s_wait_asynccnt " #n ::: "memory")

// ---- async global -> LDS (bypasses VGPRs, tracked by ASYNCcnt) ----
__device__ inline void async_b128(unsigned int ldsoff, unsigned long long gaddr) {
  asm volatile("global_load_async_to_lds_b128 %0, %1, off"
               :: "v"(ldsoff), "v"(gaddr) : "memory");
}
__device__ inline void async_b64(unsigned int ldsoff, unsigned long long gaddr) {
  asm volatile("global_load_async_to_lds_b64 %0, %1, off"
               :: "v"(ldsoff), "v"(gaddr) : "memory");
}
__device__ inline unsigned int lds_off(const void* p) {   // generic LDS ptr -> LDS byte offset
  return (unsigned int)(unsigned long long)p;
}

__device__ inline v16bf ld16(const __bf16* p) {
  v16bf r;
  uint4*       d = reinterpret_cast<uint4*>(&r);
  const uint4* s = reinterpret_cast<const uint4*>(p);
  d[0] = s[0]; d[1] = s[1];
  return r;
}

// One 16x16 n-tile of a (16 x 128) @ (128 x N) GEMM, K swept by 4 bf16 WMMAs.
__device__ inline v8f gemm_tile(const __bf16* A, const __bf16* WT, int n0, int lane) {
  const int mlo = lane & 15;
  const int kh  = lane >> 4;
  v8f acc = {};
#pragma unroll
  for (int kt = 0; kt < 4; ++kt) {
    v16bf a = ld16(A  + mlo * AS        + kt * 32 + kh * 16);
    v16bf b = ld16(WT + (n0 + mlo) * C  + kt * 32 + kh * 16);
    acc = __builtin_amdgcn_wmma_f32_16x16x32_bf16(false, a, false, b,
                                                  (short)0, acc, false, false);
  }
  return acc;
}

__device__ inline void row_ln(const float* __restrict__ x, __bf16* __restrict__ a,
                              const float* __restrict__ sc, const float* __restrict__ bi) {
  float mu = 0.f;
#pragma unroll 4
  for (int c = 0; c < C; ++c) mu += x[c];
  mu *= (1.0f / C);
  float var = 0.f;
#pragma unroll 4
  for (int c = 0; c < C; ++c) { float d = x[c] - mu; var += d * d; }
  var *= (1.0f / C);
  const float rs = rsqrtf(var + 1e-5f);
#pragma unroll 4
  for (int c = 0; c < C; ++c)
    a[c] = (__bf16)((x[c] - mu) * rs * sc[c] + bi[c]);
}

// ---------------- prep: W (K x N, f32) -> W^T (N x K, bf16) ----------------
__global__ void transpose_bf16(const float* __restrict__ src, __bf16* __restrict__ dst,
                               int K, int N) {
  int i = blockIdx.x * blockDim.x + threadIdx.x;
  if (i < K * N) {
    int n = i / K, k = i % K;
    dst[n * K + k] = (__bf16)src[k * N + n];
  }
}

// ---------------- KV tables: LN(x_kv) @ Wkv + bkv -> K/V bf16 --------------
struct KvSmem { float x[16][C]; __bf16 a[16][AS]; };

__global__ __launch_bounds__(32)
void kv_kernel(const float* __restrict__ x_kv,
               const float* __restrict__ ln_s, const float* __restrict__ ln_b,
               const __bf16* __restrict__ WkvT, const float* __restrict__ bkv,
               __bf16* __restrict__ Ktab, __bf16* __restrict__ Vtab) {
  __shared__ KvSmem sm;
  const int lane = threadIdx.x;
  const int tile = blockIdx.x;
  const int t  = tile / (NKV / 16);
  const int r0 = (tile % (NKV / 16)) * 16;
  const float* src = x_kv + ((size_t)t * NKV + r0) * C;

  // ---- Tensor Data Mover: 2D tile (tile_dim0=128 elems x tile_dim1=16 rows,
  // data_size=4B, dim0_stride=128) pulled straight into LDS. D# per ISA 8.3/8.4.
  {
    const unsigned long long ga = (unsigned long long)src;   // tile start
    u32x4 g0;
    g0[0] = 1u;                                   // count=1, user mode
    g0[1] = lds_off(&sm.x[0][0]);                 // lds_addr (bytes)
    g0[2] = (unsigned int)(ga & 0xFFFFFFFFull);   // global_addr[31:0]
    g0[3] = (unsigned int)((ga >> 32) & 0x1FFFFFFull) | (2u << 30);  // addr[56:32] | type=2
    i32x8 g1;
    g1[0] = (int)(2u << 16);                      // data_size=4B; mask/flags=0
    g1[1] = (int)(128u << 16);                    // tensor_dim0 = 128 (bits 79:48 lo)
    g1[2] = (int)(16u << 16);                     // tensor_dim0 hi=0 | tensor_dim1 = 16
    g1[3] = (int)(128u << 16);                    // tensor_dim1 hi=0 | tile_dim0 = 128
    g1[4] = 16;                                   // tile_dim1 = 16, tile_dim2 = 0
    g1[5] = 128;                                  // tensor_dim0_stride = 128
    g1[6] = 0;                                    // stride hi / dim1_stride lo
    g1[7] = 0;
    i32x4 gz4 = {0, 0, 0, 0};
    i32x8 gz8 = {0, 0, 0, 0, 0, 0, 0, 0};
    __builtin_amdgcn_tensor_load_to_lds(g0, g1, gz4, gz4, gz8, 0);
  }
  __builtin_amdgcn_s_wait_tensorcnt(0);

  if (lane < 16) row_ln(sm.x[lane], sm.a[lane], ln_s, ln_b);
  LDSF();

  const int mlo = lane & 15, hi = lane >> 4;
  for (int nt = 0; nt < 16; ++nt) {              // N = 256
    v8f acc = gemm_tile(&sm.a[0][0], WkvT, nt * 16, lane);
    const int n = nt * 16 + mlo;
    const float bb = bkv[n];
#pragma unroll
    for (int r = 0; r < 8; ++r) {
      const int m = r + hi * 8;
      const size_t row = (size_t)t * NKV + r0 + m;
      const float v = acc[r] + bb;
      if (n < C) Ktab[row * C + n]       = (__bf16)v;
      else       Vtab[row * C + (n - C)] = (__bf16)v;
    }
  }
}

// ---------------- fully fused main kernel: one s-tile per wave -------------
struct MainSmem {
  float  x[16][C];        // x_q tile, becomes residual x (f32)
  __bf16 a[16][AS];       // A staging ping
  __bf16 b[16][AS];       // A staging pong
  __bf16 kg[16][C];       // gathered K rows (9 valid, rest 0)
  __bf16 vg[16][C];       // gathered V rows
  float  sc[H][16][16];   // raw scores
  __bf16 at[H][16][32];   // attn probs, K-padded to 32
  float  bias[16];
};

__global__ __launch_bounds__(32)
void main_kernel(const float* __restrict__ x_q,
                 const int* __restrict__ nh_idx, const unsigned char* __restrict__ nh_mask,
                 const float* __restrict__ ln_q_s, const float* __restrict__ ln_q_b,
                 const __bf16* __restrict__ WqT,  const float* __restrict__ bq,
                 const __bf16* __restrict__ WaqT,
                 const __bf16* __restrict__ WoutT, const float* __restrict__ b_out,
                 const float* __restrict__ gamma,
                 const float* __restrict__ ln_m_s, const float* __restrict__ ln_m_b,
                 const __bf16* __restrict__ WmT,  const float* __restrict__ bm,
                 const __bf16* __restrict__ W1T,  const float* __restrict__ b1,
                 const __bf16* __restrict__ W2T,  const float* __restrict__ b2,
                 const float* __restrict__ gamma_mlp,
                 const __bf16* __restrict__ Ktab, const __bf16* __restrict__ Vtab,
                 float* __restrict__ out) {
  __shared__ MainSmem sm;
  const int lane = threadIdx.x;
  const int tile = blockIdx.x;
  const int t = tile / S;
  const int s = tile % S;
  const int mlo = lane & 15, hi = lane >> 4;

  // ---- issue x_q tile as 16 async b128 loads (f32 row per instruction) ----
  const float* src = x_q + ((size_t)t * NQ + (size_t)s * 16) * C;
  const unsigned int xbase = lds_off(&sm.x[0][0]);
#pragma unroll
  for (int m = 0; m < 16; ++m)
    async_b128(xbase + m * 512 + lane * 16,
               (unsigned long long)(src + m * C + lane * 4));

  // ---- issue 9 K + 9 V gather rows async (L2-resident tables) ----
  const unsigned int kbase = lds_off(&sm.kg[0][0]);
  const unsigned int vbase = lds_off(&sm.vg[0][0]);
  for (int j = 0; j < NH; ++j) {
    const int idx = nh_idx[s * NH + j];
    const size_t row = (size_t)t * NKV + idx;
    async_b64(kbase + j * 256 + lane * 8,
              (unsigned long long)(Ktab + row * C) + lane * 8);
    async_b64(vbase + j * 256 + lane * 8,
              (unsigned long long)(Vtab + row * C) + lane * 8);
  }
  // zero-pad gather rows 9..15 + mask bias (plain DS stores)
  uint2 z; z.x = 0u; z.y = 0u;
#pragma unroll
  for (int j = NH; j < 16; ++j) {
    *(uint2*)&sm.kg[j][lane * 4] = z;
    *(uint2*)&sm.vg[j][lane * 4] = z;
  }
  if (lane < 16)
    sm.bias[lane] = (lane < NH && nh_mask[s * NH + lane]) ? 0.0f : -1e9f;

  // async loads complete in order: allow the 18 gather ops to stay in flight,
  // require only the 16 x-tile rows to have landed.
  AWAITN(0x12);

  // ---- LN(x_q) -> bf16 A ----
  if (lane < 16) row_ln(sm.x[lane], sm.a[lane], ln_q_s, ln_q_b);
  LDSF();

  // ---- q = A @ Wq + bq ----
  for (int nt = 0; nt < 8; ++nt) {
    v8f acc = gemm_tile(&sm.a[0][0], WqT, nt * 16, lane);
    const int n = nt * 16 + mlo;
    const float bb = bq[n];
#pragma unroll
    for (int r = 0; r < 8; ++r) sm.b[r + hi * 8][n] = (__bf16)(acc[r] + bb);
  }
  LDSF();

  // ---- qh = q @ Wa_q (no bias) ----
  for (int nt = 0; nt < 8; ++nt) {
    v8f acc = gemm_tile(&sm.b[0][0], WaqT, nt * 16, lane);
    const int n = nt * 16 + mlo;
#pragma unroll
    for (int r = 0; r < 8; ++r) sm.a[r + hi * 8][n] = (__bf16)acc[r];
  }
  AWAIT0();   // K/V gathers hidden behind the two GEMMs above
  LDSF();

  // ---- scores[h] = q_h (16x32) @ k_h^T (32x16), one WMMA per head ----
  const float iss = 0.17677669529663687f;   // 1/sqrt(hd)
#pragma unroll
  for (int h = 0; h < H; ++h) {
    v16bf a = ld16(&sm.a[mlo][h * HD + hi * 16]);
    v16bf b = ld16(&sm.kg[mlo][h * HD + hi * 16]);
    v8f acc = {};
    acc = __builtin_amdgcn_wmma_f32_16x16x32_bf16(false, a, false, b,
                                                  (short)0, acc, false, false);
#pragma unroll
    for (int r = 0; r < 8; ++r) sm.sc[h][r + hi * 8][mlo] = acc[r] * iss;
  }
  LDSF();

  // ---- softmax over 9 neighbors (padded cols killed by -1e9 bias) ----
  for (int rr = lane; rr < 64; rr += 32) {
    const int h = rr >> 4, m = rr & 15;
    float v[16], mx = -1e30f;
#pragma unroll
    for (int n = 0; n < 16; ++n) { v[n] = sm.sc[h][m][n] + sm.bias[n]; mx = fmaxf(mx, v[n]); }
    float sum = 0.f;
#pragma unroll
    for (int n = 0; n < 16; ++n) { const float e = __expf(v[n] - mx); v[n] = e; sum += e; }
    const float inv = 1.0f / sum;
#pragma unroll
    for (int n = 0; n < 16; ++n) sm.at[h][m][n] = (__bf16)(v[n] * inv);
#pragma unroll
    for (int n = 16; n < 32; ++n) sm.at[h][m][n] = (__bf16)0.0f;
  }
  LDSF();

  // ---- attn @ V : per head two 16-wide n-tiles (K padded to 32) ----
#pragma unroll
  for (int h = 0; h < H; ++h) {
#pragma unroll
    for (int dt = 0; dt < 2; ++dt) {
      v16bf a = ld16(&sm.at[h][mlo][hi * 16]);
      v16bf b;
      __bf16* be = reinterpret_cast<__bf16*>(&b);
      const int d = dt * 16 + mlo;
      const int kb = hi * 16;
#pragma unroll
      for (int j = 0; j < 16; ++j) {
        const int k = kb + j;
        be[j] = (k < NH) ? sm.vg[k][h * HD + d] : (__bf16)0.0f;
      }
      v8f acc = {};
      acc = __builtin_amdgcn_wmma_f32_16x16x32_bf16(false, a, false, b,
                                                    (short)0, acc, false, false);
#pragma unroll
      for (int r = 0; r < 8; ++r) sm.b[r + hi * 8][h * HD + d] = (__bf16)acc[r];
    }
  }
  LDSF();

  // ---- x = x_q + gamma * (attn_out @ W_out + b_out) ----
  for (int nt = 0; nt < 8; ++nt) {
    v8f acc = gemm_tile(&sm.b[0][0], WoutT, nt * 16, lane);
    const int n = nt * 16 + mlo;
    const float bo = b_out[n], g = gamma[n];
#pragma unroll
    for (int r = 0; r < 8; ++r) {
      const int m = r + hi * 8;
      sm.x[m][n] += g * (acc[r] + bo);
    }
  }
  LDSF();

  // ---- LN(x) -> A ----
  if (lane < 16) row_ln(sm.x[lane], sm.a[lane], ln_m_s, ln_m_b);
  LDSF();

  // ---- h = A @ Wm + bm ----
  for (int nt = 0; nt < 8; ++nt) {
    v8f acc = gemm_tile(&sm.a[0][0], WmT, nt * 16, lane);
    const int n = nt * 16 + mlo;
    const float bb = bm[n];
#pragma unroll
    for (int r = 0; r < 8; ++r) sm.b[r + hi * 8][n] = (__bf16)(acc[r] + bb);
  }
  LDSF();

  // ---- g = gelu(h @ W1 + b1)  (tanh approximation, jax default) ----
  for (int nt = 0; nt < 8; ++nt) {
    v8f acc = gemm_tile(&sm.b[0][0], W1T, nt * 16, lane);
    const int n = nt * 16 + mlo;
    const float bb = b1[n];
#pragma unroll
    for (int r = 0; r < 8; ++r) {
      const float hv = acc[r] + bb;
      const float gv = 0.5f * hv *
          (1.0f + tanhf(0.7978845608028654f * (hv + 0.044715f * hv * hv * hv)));
      sm.a[r + hi * 8][n] = (__bf16)gv;
    }
  }
  LDSF();

  // ---- final = x + gamma_mlp * (g @ W2 + b2), streamed to HBM ----
  float* dst = out + ((size_t)t * NQ + (size_t)s * 16) * C;
  for (int nt = 0; nt < 8; ++nt) {
    v8f acc = gemm_tile(&sm.a[0][0], W2T, nt * 16, lane);
    const int n = nt * 16 + mlo;
    const float bb = b2[n], gm = gamma_mlp[n];
#pragma unroll
    for (int r = 0; r < 8; ++r) {
      const int m = r + hi * 8;
      dst[(size_t)m * C + n] = sm.x[m][n] + gm * (acc[r] + bb);
    }
  }
}

// ---------------------------------------------------------------------------
extern "C" void kernel_launch(void* const* d_in, const int* in_sizes, int n_in,
                              void* d_out, int out_size, void* d_ws, size_t ws_size,
                              hipStream_t stream) {
  const float* x_q    = (const float*)d_in[0];
  const float* x_kv   = (const float*)d_in[1];
  const int*   nh_idx = (const int*)d_in[2];
  const unsigned char* nh_mask = (const unsigned char*)d_in[3];
  const float* ln_q_s = (const float*)d_in[4];
  const float* ln_q_b = (const float*)d_in[5];
  const float* Wq     = (const float*)d_in[6];
  const float* bq     = (const float*)d_in[7];
  const float* ln_kv_s= (const float*)d_in[8];
  const float* ln_kv_b= (const float*)d_in[9];
  const float* Wkv    = (const float*)d_in[10];
  const float* bkv    = (const float*)d_in[11];
  const float* Wa_q   = (const float*)d_in[12];
  const float* W_out  = (const float*)d_in[13];
  const float* b_out  = (const float*)d_in[14];
  const float* gamma  = (const float*)d_in[15];
  const float* ln_m_s = (const float*)d_in[16];
  const float* ln_m_b = (const float*)d_in[17];
  const float* Wm     = (const float*)d_in[18];
  const float* bm     = (const float*)d_in[19];
  const float* W1     = (const float*)d_in[20];
  const float* b1     = (const float*)d_in[21];
  const float* W2     = (const float*)d_in[22];
  const float* b2     = (const float*)d_in[23];
  const float* gamma_mlp = (const float*)d_in[24];
  float* out = (float*)d_out;

  // workspace layout (bf16), byte offsets
  char* wsb = (char*)d_ws;
  __bf16* WqT   = (__bf16*)(wsb + 0 * 32768);
  __bf16* WaqT  = (__bf16*)(wsb + 1 * 32768);
  __bf16* WoutT = (__bf16*)(wsb + 2 * 32768);
  __bf16* WmT   = (__bf16*)(wsb + 3 * 32768);
  __bf16* W1T   = (__bf16*)(wsb + 4 * 32768);
  __bf16* W2T   = (__bf16*)(wsb + 5 * 32768);
  __bf16* WkvT  = (__bf16*)(wsb + 6 * 32768);                // 256 x 128
  __bf16* Ktab  = (__bf16*)(wsb + 6 * 32768 + 65536);        // T*NKV*C bf16
  __bf16* Vtab  = Ktab + (size_t)T * NKV * C;

  // 1) weight transpose+downconvert (tiny; L2-resident thereafter)
  transpose_bf16<<<64, 256, 0, stream>>>(Wq,    WqT,   128, 128);
  transpose_bf16<<<64, 256, 0, stream>>>(Wa_q,  WaqT,  128, 128);
  transpose_bf16<<<64, 256, 0, stream>>>(W_out, WoutT, 128, 128);
  transpose_bf16<<<64, 256, 0, stream>>>(Wm,    WmT,   128, 128);
  transpose_bf16<<<64, 256, 0, stream>>>(W1,    W1T,   128, 128);
  transpose_bf16<<<64, 256, 0, stream>>>(W2,    W2T,   128, 128);
  transpose_bf16<<<128, 256, 0, stream>>>(Wkv,  WkvT,  128, 256);

  // 2) K/V tables (12.6 MB bf16 -> L2-resident for the gather phase)
  kv_kernel<<<T * NKV / 16, 32, 0, stream>>>(x_kv, ln_kv_s, ln_kv_b, WkvT, bkv,
                                             Ktab, Vtab);

  // 3) fully fused attention + MLP, one 16-query s-tile per wave
  main_kernel<<<T * S, 32, 0, stream>>>(x_q, nh_idx, nh_mask,
                                        ln_q_s, ln_q_b, WqT, bq, WaqT,
                                        WoutT, b_out, gamma,
                                        ln_m_s, ln_m_b, WmT, bm,
                                        W1T, b1, W2T, b2, gamma_mlp,
                                        Ktab, Vtab, out);
  (void)in_sizes; (void)n_in; (void)out_size; (void)ws_size;
}
